// FastEnsembleDeepSDFMirrored_46505905881200
// MI455X (gfx1250) — matrix operand order repaired and checked by reading, not compiled, not run on hardware
//
#include <hip/hip_runtime.h>
#include <math.h>

// ---------------- problem constants ----------------
#define E_NUM   40
#define NSYM    16
#define NKPS    39
#define NPTS    4096
#define LATS    1344          // GLOB + E*LOC
#define GLOBD   64
#define LOCD    32
#define TILE    32            // points per workgroup (2 waves x 16 rows)
#define INV_SQRT2 0.70710678118654752440f

// packed-weight layout (elements). per layer: opad x kpad bf16, row-major over o.
//  l:      0      1      2      3      4      5      6      7      8
// di:     99    200    200    200    200    200    200    200    200
// do:    200    200    200    101    200    200    200    200      1
// kpad:  128    224    224    224    224    224    224    224    224
// opad:  208    208    208    112    208    208    208    208     16
#define WPM 334848            // weight elems per member
#define BPM 1584              // bias elems per member
#define WOFF0 0
#define WOFF1 26624
#define WOFF2 73216
#define WOFF3 119808
#define WOFF4 144896
#define WOFF5 191488
#define WOFF6 238080
#define WOFF7 284672
#define WOFF8 331264
#define BOFF0 0
#define BOFF1 208
#define BOFF2 416
#define BOFF3 624
#define BOFF4 736
#define BOFF5 944
#define BOFF6 1152
#define BOFF7 1360
#define BOFF8 1568

// workspace byte offsets
#define OFF_ANCH 0u
#define OFF_BIAS 512u
#define OFF_SDF  (OFF_BIAS + (size_t)E_NUM * BPM * 4)             // 40*1584 floats
#define OFF_W    (OFF_SDF + (size_t)E_NUM * NPTS * 4)             // 40*4096 floats

// ---------------- types ----------------
typedef __attribute__((ext_vector_type(16))) __bf16          v16bf;
typedef __attribute__((ext_vector_type(8)))  float           v8f;
typedef __attribute__((ext_vector_type(8)))  unsigned short  v8us;
typedef __attribute__((ext_vector_type(16))) unsigned short  v16us;

// ---------------- helpers ----------------
__device__ __forceinline__ unsigned short f2bf(float f) {
  unsigned int u = __float_as_uint(f);
  u += 0x7FFFu + ((u >> 16) & 1u);              // round-to-nearest-even
  return (unsigned short)(u >> 16);
}
__device__ __forceinline__ float bf2f(unsigned short h) {
  return __uint_as_float(((unsigned int)h) << 16);
}
// softplus(100x)/100 via hardware transcendentals (v_exp_f32 / v_log_f32).
// For y<=20, 1+exp(y) <= 4.9e8 is well inside fp32 range; y>20 -> identity.
__device__ __forceinline__ float softplus100(float x) {
  float y = 100.f * x;
  float e = __expf(y);
  float s = __logf(1.f + e) * 0.01f;
  return (y > 20.f) ? x : s;
}

// A fragment: 16x32 bf16, rows = points.  Lanes 0-15: K=k0..k0+7 then k0+16..k0+23.
// Lanes 16-31: K=k0+8..k0+15 then k0+24..k0+31.  (ISA 7.12.2 16-bit A layout)
__device__ __forceinline__ v16bf load_fragA(const unsigned short* base, int stride,
                                            int row0, int k0, int lane) {
  const int half = lane >> 4, m = lane & 15;
  const unsigned short* p = base + (row0 + m) * stride + k0 + half * 8;
  union { v16us v; v8us h[2]; } u;
  u.h[0] = *(const v8us*)(p);
  u.h[1] = *(const v8us*)(p + 16);
  return __builtin_bit_cast(v16bf, u.v);
}
// B fragment: 32x16 bf16, lane selects column N; lanes 0-15 hold K=k0..k0+7 then k0+16..k0+23,
// lanes 16-31 hold K=k0+8..k0+15 then k0+24..k0+31 (same K order as A).
__device__ __forceinline__ v16bf load_fragB(const unsigned short* base, int stride,
                                            int col0, int k0, int lane) {
  const int half = lane >> 4, n = lane & 15;
  const unsigned short* p = base + (col0 + n) * stride + k0 + half * 8;
  union { v16us v; v8us h[2]; } u;
  u.h[0] = *(const v8us*)(p);
  u.h[1] = *(const v8us*)(p + 16);
  return __builtin_bit_cast(v16bf, u.v);
}

// One K-pass over NT 16-wide output tiles starting at column col0, for one
// wave's 16 point-rows.  NT kept small (<=7) so accumulators (NT*8 VGPRs)
// plus pipelined B fragments fit comfortably in 256 VGPRs.
template <int NT>
__device__ __forceinline__ void layer_gemm(
    const unsigned short* __restrict__ W,    // global bf16, opad x kp
    int kp,                                  // K stride (padded, mult of 32)
    const float* __restrict__ bias,          // packed fp32, length opad
    const unsigned short* __restrict__ actin,// LDS bf16, stride kp
    unsigned short* __restrict__ actout,     // LDS bf16, stride kpn (or null)
    int kpn, int col0, int dout, bool do_sp, float scale,
    float* __restrict__ gout,                // if non-null: write column 0 here
    int m_base, int lane) {
  const int half = lane >> 4;
  const int nn   = lane & 15;
  v8f acc[NT];
  v8f zero = {0.f, 0.f, 0.f, 0.f, 0.f, 0.f, 0.f, 0.f};
#pragma unroll
  for (int t = 0; t < NT; ++t) acc[t] = zero;

  for (int k0 = 0; k0 < kp; k0 += 32) {
    v16bf a = load_fragA(actin, kp, m_base, k0, lane);
#pragma unroll
    for (int t = 0; t < NT; ++t) {
      v16bf b = load_fragB(W, kp, col0 + t * 16, k0, lane);
      acc[t] = __builtin_amdgcn_wmma_f32_16x16x32_bf16(
          false, a, false, b, (short)0, acc[t], false, false);
    }
  }

#pragma unroll
  for (int t = 0; t < NT; ++t) {
    const int o = col0 + t * 16 + nn;
    const float bo = bias[o];
#pragma unroll
    for (int r = 0; r < 8; ++r) {
      float v = acc[t][r] + bo;
      if (do_sp) v = softplus100(v);
      v *= scale;
      if (o >= dout) v = 0.f;                 // keep pad columns exactly zero
      const int row = m_base + r + half * 8;  // C layout: VGPR r -> M=r / r+8
      if (gout) {
        if (o == 0) gout[row] = v;
      } else {
        actout[row * kpn + o] = f2bf(v);
      }
    }
  }
}

// Full 208-wide layer = two passes (7 tiles + 6 tiles) to halve register load.
__device__ __forceinline__ void layer_full(
    const unsigned short* __restrict__ W, int kp,
    const float* __restrict__ bias,
    const unsigned short* __restrict__ actin,
    unsigned short* __restrict__ actout, int kpn,
    int dout, bool do_sp, float scale, int m_base, int lane) {
  layer_gemm<7>(W, kp, bias, actin, actout, kpn, 0,   dout, do_sp, scale,
                nullptr, m_base, lane);
  layer_gemm<6>(W, kp, bias, actin, actout, kpn, 112, dout, do_sp, scale,
                nullptr, m_base, lane);
}

// ---------------- kernel 1: anchor pre-net ----------------
__global__ __launch_bounds__(256) void prenet_kernel(
    const float* __restrict__ lat, const float* __restrict__ anchors_const,
    const float* __restrict__ pw0, const float* __restrict__ pb0,
    const float* __restrict__ pw1, const float* __restrict__ pb1,
    const float* __restrict__ pw2, const float* __restrict__ pb2,
    float* __restrict__ anch_ws, float* __restrict__ out_tail) {
  __shared__ float g[GLOBD];
  __shared__ float h0[256];
  __shared__ float h1[256];
  const int t = threadIdx.x;
  if (t < GLOBD) g[t] = lat[t];               // lat_rep[0,0,:64]
  __syncthreads();
  {
    float a = pb0[t];
    for (int i = 0; i < GLOBD; ++i) a += g[i] * pw0[i * 256 + t];
    h0[t] = fmaxf(a, 0.f);
  }
  __syncthreads();
  {
    float a = pb1[t];
    for (int i = 0; i < 256; ++i) a += h0[i] * pw1[i * 256 + t];
    h1[t] = fmaxf(a, 0.f);
  }
  __syncthreads();
  if (t < NKPS * 3) {
    float a = pb2[t];
    for (int i = 0; i < 256; ++i) a += h1[i] * pw2[i * (NKPS * 3) + t];
    a += anchors_const[t];
    anch_ws[t]  = a;
    out_tail[t] = a;
  }
}

// ---------------- kernel 2: weight packing (per layer) ----------------
__global__ __launch_bounds__(256) void pack_weights_kernel(
    const float* __restrict__ ew, const float* __restrict__ eb,
    unsigned short* __restrict__ wdst, float* __restrict__ bdst,
    int di, int dd, int kp, int op, long woff, long boff) {
  const int e = blockIdx.x;
  const int s = (e < 2 * NSYM) ? (e >> 1) : (e - NSYM);  // _expand mapping
  unsigned short* wd = wdst + (size_t)e * WPM + woff;
  const float* src = ew + (size_t)s * dd * di;
  const int tot = op * kp;
  for (int idx = threadIdx.x; idx < tot; idx += blockDim.x) {
    const int o = idx / kp, k = idx - o * kp;
    const float v = (o < dd && k < di) ? src[o * di + k] : 0.f;
    wd[idx] = f2bf(v);
  }
  float* bd = bdst + (size_t)e * BPM + boff;
  for (int idx = threadIdx.x; idx < op; idx += blockDim.x)
    bd[idx] = (idx < dd) ? eb[(size_t)s * dd + idx] : 0.f;
}

// ---------------- kernel 3: ensemble MLP (WMMA) ----------------
__global__ __launch_bounds__(64) void ensemble_mlp_kernel(
    const float* __restrict__ xyz, const float* __restrict__ lat,
    const float* __restrict__ anch_ws,
    const unsigned short* __restrict__ Wp, const float* __restrict__ Bp,
    float* __restrict__ sdf_ws) {
  __shared__ unsigned short sX[TILE * 128];   // input (99 -> pad 128)
  __shared__ unsigned short sA[TILE * 224];   // ping
  __shared__ unsigned short sB[TILE * 224];   // pong
  const int e    = blockIdx.y;
  const int n0   = blockIdx.x * TILE;
  const int tid  = threadIdx.x;
  const int lane = tid & 31;
  const int m_base = (tid >> 5) * 16;

  float ax = 0.f, ay = 0.f, az = 0.f;
  if (e < NKPS) { ax = anch_ws[e * 3]; ay = anch_ws[e * 3 + 1]; az = anch_ws[e * 3 + 2]; }
  const float mir = (e < 2 * NSYM && (e & 1)) ? -1.f : 1.f;

  // zero activation buffers once (pad regions must stay zero)
  for (int i = tid; i < TILE * 224; i += 64) { sA[i] = 0; sB[i] = 0; }

  // build bf16 input tile: [cx*mir, cy, cz, g(64), local(32), 0-pad]
  for (int idx = tid; idx < TILE * 128; idx += 64) {
    const int p = idx >> 7, c = idx & 127;
    const int n = n0 + p;
    float v = 0.f;
    if (c == 0)       v = (xyz[n * 3 + 0] - ax) * mir;
    else if (c == 1)  v =  xyz[n * 3 + 1] - ay;
    else if (c == 2)  v =  xyz[n * 3 + 2] - az;
    else if (c < 3 + GLOBD)          v = lat[(size_t)n * LATS + (c - 3)];
    else if (c < 3 + GLOBD + LOCD)   v = lat[(size_t)n * LATS + GLOBD + e * LOCD + (c - 3 - GLOBD)];
    sX[p * 128 + c] = f2bf(v);
  }
  __syncthreads();

  const unsigned short* We = Wp + (size_t)e * WPM;
  const float* Be = Bp + (size_t)e * BPM;
  float* gout = sdf_ws + (size_t)e * NPTS + n0;
  __builtin_prefetch(We, 0, 1);

  layer_full(We + WOFF0, 128, Be + BOFF0, sX, sA, 224, 200, true, 1.f, m_base, lane);
  layer_full(We + WOFF1, 224, Be + BOFF1, sA, sB, 224, 200, true, 1.f, m_base, lane);
  layer_full(We + WOFF2, 224, Be + BOFF2, sB, sA, 224, 200, true, 1.f, m_base, lane);
  // layer 3: 101 outputs (7 tiles), scaled by 1/sqrt(2) for the skip concat
  layer_gemm<7>(We + WOFF3, 224, Be + BOFF3, sA, sB, 224, 0, 101, true, INV_SQRT2,
                nullptr, m_base, lane);
  // skip: cols [101,200) = input * 1/sqrt(2); per-wave rows -> no barrier needed
  for (int idx = lane; idx < 16 * 99; idx += 32) {
    const int r = idx / 99, c = idx - r * 99;
    const int row = m_base + r;
    sB[row * 224 + 101 + c] = f2bf(bf2f(sX[row * 128 + c]) * INV_SQRT2);
  }
  layer_full(We + WOFF4, 224, Be + BOFF4, sB, sA, 224, 200, true, 1.f, m_base, lane);
  layer_full(We + WOFF5, 224, Be + BOFF5, sA, sB, 224, 200, true, 1.f, m_base, lane);
  layer_full(We + WOFF6, 224, Be + BOFF6, sB, sA, 224, 200, true, 1.f, m_base, lane);
  layer_full(We + WOFF7, 224, Be + BOFF7, sA, sB, 224, 200, true, 1.f, m_base, lane);
  layer_gemm<1>(We + WOFF8, 224, Be + BOFF8, sB, nullptr, 0, 0, 1, false, 1.f,
                gout, m_base, lane);
}

// ---------------- kernel 4: anchor-weighted blend ----------------
__global__ __launch_bounds__(256) void blend_kernel(
    const float* __restrict__ xyz, const float* __restrict__ anch_ws,
    const float* __restrict__ sdf_ws, float* __restrict__ out) {
  __shared__ float sAnch[NKPS * 3];
  const int tid = threadIdx.x;
  if (tid < NKPS * 3) sAnch[tid] = anch_ws[tid];
  __syncthreads();
  const int n = blockIdx.x * blockDim.x + tid;
  if (n >= NPTS) return;
  const float x = xyz[n * 3], y = xyz[n * 3 + 1], z = xyz[n * 3 + 2];
  float wsum = 1e-6f, acc = 0.f;
  for (int e = 0; e < E_NUM; ++e) {
    float d;
    if (e < NKPS) {
      const float dx = sAnch[e * 3] - x, dy = sAnch[e * 3 + 1] - y, dz = sAnch[e * 3 + 2] - z;
      const float nrm = sqrtf(dx * dx + dy * dy + dz * dz) + 1e-5f;
      d = -(nrm * nrm);
    } else {
      d = -0.2f;
    }
    const float w = __expf(d * 100.f);
    wsum += w;
    acc  += w * sdf_ws[(size_t)e * NPTS + n];
  }
  out[n] = acc / wsum;
}

// ---------------- host launcher ----------------
extern "C" void kernel_launch(void* const* d_in, const int* in_sizes, int n_in,
                              void* d_out, int out_size, void* d_ws, size_t ws_size,
                              hipStream_t stream) {
  (void)in_sizes; (void)n_in; (void)out_size; (void)ws_size;
  const float* xyz = (const float*)d_in[0];
  const float* lat = (const float*)d_in[1];
  const float* anc = (const float*)d_in[2];
  const float* pw0 = (const float*)d_in[3];
  const float* pb0 = (const float*)d_in[4];
  const float* pw1 = (const float*)d_in[5];
  const float* pb1 = (const float*)d_in[6];
  const float* pw2 = (const float*)d_in[7];
  const float* pb2 = (const float*)d_in[8];

  float* out = (float*)d_out;
  char*  ws  = (char*)d_ws;
  float*          anch_ws = (float*)(ws + OFF_ANCH);
  float*          bias_ws = (float*)(ws + OFF_BIAS);
  float*          sdf_ws  = (float*)(ws + OFF_SDF);
  unsigned short* w_ws    = (unsigned short*)(ws + OFF_W);

  prenet_kernel<<<1, 256, 0, stream>>>(lat, anc, pw0, pb0, pw1, pb1, pw2, pb2,
                                       anch_ws, out + NPTS);

  static const int  DI[9] = {99, 200, 200, 200, 200, 200, 200, 200, 200};
  static const int  DO[9] = {200, 200, 200, 101, 200, 200, 200, 200, 1};
  static const int  KP[9] = {128, 224, 224, 224, 224, 224, 224, 224, 224};
  static const int  OP[9] = {208, 208, 208, 112, 208, 208, 208, 208, 16};
  static const long WF[9] = {WOFF0, WOFF1, WOFF2, WOFF3, WOFF4, WOFF5, WOFF6, WOFF7, WOFF8};
  static const long BF[9] = {BOFF0, BOFF1, BOFF2, BOFF3, BOFF4, BOFF5, BOFF6, BOFF7, BOFF8};
  for (int l = 0; l < 9; ++l) {
    pack_weights_kernel<<<E_NUM, 256, 0, stream>>>(
        (const float*)d_in[9 + 2 * l], (const float*)d_in[10 + 2 * l],
        w_ws, bias_ws, DI[l], DO[l], KP[l], OP[l], WF[l], BF[l]);
  }

  dim3 gridM(NPTS / TILE, E_NUM);
  ensemble_mlp_kernel<<<gridM, 64, 0, stream>>>(xyz, lat, anch_ws, w_ws, bias_ws, sdf_ws);

  blend_kernel<<<NPTS / 256, 256, 0, stream>>>(xyz, anch_ws, sdf_ws, out);
}
